// Qwen3SparseMoeBlock_17583596110548
// MI455X (gfx1250) — matrix operand order repaired and checked by reading, not compiled, not run on hardware
//
#include <hip/hip_runtime.h>
#include <hip/hip_bf16.h>
#include <cstdint>
#include <cstddef>

// ---- problem constants (mirror reference) ----
#define HDIM   1024
#define IDIM   512
#define ISDIM  2048
#define NEXP   16
#define TOPK   4
#define TTOK   1024   // B*S

typedef uint32_t u32;
typedef __attribute__((ext_vector_type(16))) __bf16 v16bf;
typedef __attribute__((ext_vector_type(8)))  __bf16 v8bf;
typedef __attribute__((ext_vector_type(8)))  float  v8f;

__device__ __forceinline__ v8f wmma_bf16(v16bf a, v16bf b, v8f c) {
  return __builtin_amdgcn_wmma_f32_16x16x32_bf16(false, a, false, b, (short)0, c,
                                                 false, false);
}

__device__ __forceinline__ v16bf pack16(v8bf lo, v8bf hi) {
  v16bf r;
#pragma unroll
  for (int j = 0; j < 8; ++j) { r[j] = lo[j]; r[j + 8] = hi[j]; }
  return r;
}

__device__ __forceinline__ float fast_rcp(float x) {
#if __has_builtin(__builtin_amdgcn_rcpf)
  return __builtin_amdgcn_rcpf(x);
#else
  return 1.f / x;
#endif
}

// ---- CDNA5 async global->LDS copy (ASYNCcnt-tracked DMA) ----
__device__ __forceinline__ void async_copy_b128(u32 lds_byte_off, const void* gaddr) {
  asm volatile("global_load_async_to_lds_b128 %0, %1, off"
               :: "v"(lds_byte_off), "v"(gaddr)
               : "memory");
}
template <int N>
__device__ __forceinline__ void wait_async() {
  asm volatile("s_wait_asynccnt %0" :: "n"(N) : "memory");
}

// Stage a 32(K) x 16(N) f32 tile (row stride ldb floats) into LDS.
// LDS slot layout (528 floats / 2112 B): rows 0-15 at [k*16+c],
// rows 16-31 at [272 + (k-16)*16 + c]  (64B skew -> conflict-free frag reads).
__device__ __forceinline__ void stage_tile(u32 lds_base, const float* src, int ldb) {
  const int l = threadIdx.x & 31;
#pragma unroll
  for (int i = 0; i < 4; ++i) {
    const int r = (l >> 2) + i * 8;          // K row
    const int c = (l & 3) * 4;               // N col (4 floats / 16B per lane)
    const u32 idx = (u32)((r >> 4) * 272 + (r & 15) * 16 + c);
    async_copy_b128(lds_base + idx * 4u, src + (size_t)r * ldb + c);
  }
}

// Build the 16x16x32 B fragment (lane: N = l&15, K = (j>>3)*16 + (l>>4)*8 + (j&7)) from a
// staged f32 tile in LDS, converting to bf16 in-register.
__device__ __forceinline__ v16bf bfrag(const float* tile) {
  const int l = threadIdx.x & 31;
  const int col = l & 15, hi = l >> 4;
  v16bf B;
#pragma unroll
  for (int j = 0; j < 16; ++j) B[j] = (__bf16)tile[(j >> 3) * 272 + (hi * 8 + (j & 7)) * 16 + col];
  return B;
}

// ---------------- kernel 1: x -> bf16 ----------------
__global__ void convert_x_kernel(const float* __restrict__ x, __bf16* __restrict__ xb) {
  int i = blockIdx.x * blockDim.x + threadIdx.x;
  if (i < TTOK * HDIM) xb[i] = (__bf16)x[i];
}

// ---------------- kernel 2: router + topk + shared gate ----------------
__global__ void router_kernel(const float* __restrict__ x,
                              const float* __restrict__ w_router,
                              const float* __restrict__ w_sg,
                              float* __restrict__ logits_out,
                              int* __restrict__ top_i,
                              float* __restrict__ top_w,
                              float* __restrict__ sgate) {
  __shared__ float part[NEXP + 1][16];
  const int t   = blockIdx.x;
  const int tid = threadIdx.x;          // 256 threads
  const float* xr = x + (size_t)t * HDIM;

  const int e = tid >> 4, l16 = tid & 15;
  float acc = 0.f;
  for (int h = l16; h < HDIM; h += 16) acc += xr[h] * w_router[h * NEXP + e];
  part[e][l16] = acc;
  if (tid < 16) {                        // shared-expert gate dot
    float a2 = 0.f;
    for (int h = tid; h < HDIM; h += 16) a2 += xr[h] * w_sg[h];
    part[NEXP][tid] = a2;
  }
  __syncthreads();

  if (tid == 0) {
    float lg[NEXP];
    for (int ee = 0; ee < NEXP; ++ee) {
      float s = 0.f;
      for (int j = 0; j < 16; ++j) s += part[ee][j];
      lg[ee] = s;
      logits_out[(size_t)t * NEXP + ee] = s;     // raw logits output
    }
    float sg = 0.f;
    for (int j = 0; j < 16; ++j) sg += part[NEXP][j];
    sgate[t] = fast_rcp(1.f + __expf(-sg));

    float mx = lg[0];
    for (int ee = 1; ee < NEXP; ++ee) mx = fmaxf(mx, lg[ee]);
    float p[NEXP], sum = 0.f;
    for (int ee = 0; ee < NEXP; ++ee) { p[ee] = __expf(lg[ee] - mx); sum += p[ee]; }
    float rs = fast_rcp(sum);
    for (int ee = 0; ee < NEXP; ++ee) p[ee] *= rs;

    int   idx[TOPK]; float wv[TOPK]; float tsum = 0.f;
    for (int k = 0; k < TOPK; ++k) {
      int am = 0; float bv = -1.f;
      for (int ee = 0; ee < NEXP; ++ee)
        if (p[ee] > bv) { bv = p[ee]; am = ee; }
      idx[k] = am; wv[k] = bv; p[am] = -2.f; tsum += bv;
    }
    float rt = fast_rcp(tsum);
    for (int k = 0; k < TOPK; ++k) {
      top_i[t * TOPK + k] = idx[k];
      top_w[t * TOPK + k] = wv[k] * rt;
    }
  }
}

// ---------------- kernel 3: per-expert token lists (deterministic) ----------------
__global__ void build_lists_kernel(const int* __restrict__ top_i,
                                   int* __restrict__ lists,
                                   int* __restrict__ counts) {
  int e = threadIdx.x;
  if (e >= NEXP) return;
  int n = 0;
  for (int t = 0; t < TTOK; ++t)
    for (int k = 0; k < TOPK; ++k)
      if (top_i[t * TOPK + k] == e)
        lists[e * TTOK + n++] = t | (k << 20);
  counts[e] = n;
}

// ---------------- kernel 4: grouped gate/up GEMM + SwiGLU -> hbuf(bf16) ----------------
// One wave per block; 64(M) x 16(N) tile; B tiles double-buffered via async DMA.
__global__ __launch_bounds__(32)
void expert_gateup_kernel(const __bf16* __restrict__ xb,
                          const float* __restrict__ w_gate,
                          const float* __restrict__ w_up,
                          const int* __restrict__ lists,
                          const int* __restrict__ counts,
                          __bf16* __restrict__ hbuf) {
  __shared__ float stage[4][528];                 // [buf*2+mat]
  const int COLT = IDIM / 16, ROWB = TTOK / 64;
  int bid = blockIdx.x;
  const int ct = bid % COLT; bid /= COLT;
  const int rb = bid % ROWB; const int e = bid / ROWB;
  const int cnt = counts[e];
  const int m0  = rb * 64;
  if (m0 >= cnt) return;

  const int l  = threadIdx.x;
  const int hi = l >> 4;
  const int n  = ct * 16 + (l & 15);
  const float* wgt = w_gate + (size_t)e * HDIM * IDIM + ct * 16;
  const float* wut = w_up   + (size_t)e * HDIM * IDIM + ct * 16;

  const __bf16* xrows[4];
#pragma unroll
  for (int s = 0; s < 4; ++s) {
    int mr = m0 + s * 16 + (l & 15);
    if (mr >= cnt) mr = cnt - 1;                 // pad row (results discarded)
    xrows[s] = xb + (size_t)(lists[e * TTOK + mr] & 0xFFFF) * HDIM;
  }

  const u32 lb = (u32)(uintptr_t)&stage[0][0];
  stage_tile(lb + 0u * 2112u, wgt, IDIM);        // prologue: buf 0
  stage_tile(lb + 1u * 2112u, wut, IDIM);

  v8f aG[4] = {}, aU[4] = {};
  int buf = 0;
  for (int k0 = 0; k0 < HDIM; k0 += 32) {
    const int kn = (k0 + 32 < HDIM) ? k0 + 32 : 0;   // clamped prefetch addr
    stage_tile(lb + (u32)((buf ^ 1) * 2 + 0) * 2112u, wgt + (size_t)kn * IDIM, IDIM);
    stage_tile(lb + (u32)((buf ^ 1) * 2 + 1) * 2112u, wut + (size_t)kn * IDIM, IDIM);
    wait_async<8>();                             // current buffer complete
    v16bf Bg = bfrag(&stage[buf * 2 + 0][0]);
    v16bf Bu = bfrag(&stage[buf * 2 + 1][0]);
    const int ka = k0 + hi * 8;
#pragma unroll
    for (int s = 0; s < 4; ++s) {
      v16bf A = pack16(*(const v8bf*)(xrows[s] + ka), *(const v8bf*)(xrows[s] + ka + 16));
      aG[s] = wmma_bf16(A, Bg, aG[s]);
      aU[s] = wmma_bf16(A, Bu, aU[s]);
    }
    buf ^= 1;
  }
#pragma unroll
  for (int s = 0; s < 4; ++s)
#pragma unroll
    for (int v = 0; v < 8; ++v) {
      const int gr = m0 + s * 16 + v + 8 * hi;   // < TTOK always; pad rows never read
      float g = aG[s][v], u = aU[s][v];
      float h = g * fast_rcp(1.f + __expf(-g)) * u;
      hbuf[((size_t)e * TTOK + gr) * IDIM + n] = (__bf16)h;
    }
}

// ---------------- kernel 5: grouped down GEMM -> weighted partial [T,4,H] ----------------
__global__ __launch_bounds__(32)
void expert_down_kernel(const __bf16* __restrict__ hbuf,
                        const float* __restrict__ w_down,
                        const int* __restrict__ lists,
                        const int* __restrict__ counts,
                        const float* __restrict__ top_w,
                        float* __restrict__ partial) {
  __shared__ float stage[2][528];
  const int COLT = HDIM / 16, ROWB = TTOK / 64;
  int bid = blockIdx.x;
  const int ct = bid % COLT; bid /= COLT;
  const int rb = bid % ROWB; const int e = bid / ROWB;
  const int cnt = counts[e];
  const int m0  = rb * 64;
  if (m0 >= cnt) return;

  const int l  = threadIdx.x;
  const int hi = l >> 4;
  const int n  = ct * 16 + (l & 15);
  const float* wdt = w_down + (size_t)e * IDIM * HDIM + ct * 16;

  const __bf16* arows[4];
#pragma unroll
  for (int s = 0; s < 4; ++s) {
    int mr = m0 + s * 16 + (l & 15);
    if (mr >= cnt) mr = cnt - 1;
    arows[s] = hbuf + ((size_t)e * TTOK + mr) * IDIM;
  }

  const u32 lb = (u32)(uintptr_t)&stage[0][0];
  stage_tile(lb, wdt, HDIM);

  v8f acc[4] = {};
  int buf = 0;
  for (int k0 = 0; k0 < IDIM; k0 += 32) {
    const int kn = (k0 + 32 < IDIM) ? k0 + 32 : 0;
    stage_tile(lb + (u32)(buf ^ 1) * 2112u, wdt + (size_t)kn * HDIM, HDIM);
    wait_async<4>();
    v16bf B = bfrag(&stage[buf][0]);
    const int ka = k0 + hi * 8;
#pragma unroll
    for (int s = 0; s < 4; ++s) {
      v16bf A = pack16(*(const v8bf*)(arows[s] + ka), *(const v8bf*)(arows[s] + ka + 16));
      acc[s] = wmma_bf16(A, B, acc[s]);
    }
    buf ^= 1;
  }
#pragma unroll
  for (int s = 0; s < 4; ++s)
#pragma unroll
    for (int v = 0; v < 8; ++v) {
      const int gr = m0 + s * 16 + v + 8 * hi;
      if (gr < cnt) {
        const int li = lists[e * TTOK + gr];
        const int t  = li & 0xFFFF;
        const int k  = li >> 20;
        partial[((size_t)t * TOPK + k) * HDIM + n] = top_w[t * TOPK + k] * acc[s][v];
      }
    }
}

// ---------------- kernel 6: shared expert gate/up -> hs(bf16) ----------------
__global__ __launch_bounds__(32)
void shared_gateup_kernel(const __bf16* __restrict__ xb,
                          const float* __restrict__ ws_gate,
                          const float* __restrict__ ws_up,
                          __bf16* __restrict__ hs) {
  __shared__ float stage[4][528];
  const int COLT = ISDIM / 16;
  const int ct = blockIdx.x % COLT;
  const int rb = blockIdx.x / COLT;
  const int l  = threadIdx.x;
  const int hi = l >> 4;
  const int n  = ct * 16 + (l & 15);
  const float* wgt = ws_gate + ct * 16;
  const float* wut = ws_up   + ct * 16;

  const __bf16* xrows[4];
#pragma unroll
  for (int s = 0; s < 4; ++s)
    xrows[s] = xb + (size_t)(rb * 64 + s * 16 + (l & 15)) * HDIM;

  const u32 lb = (u32)(uintptr_t)&stage[0][0];
  stage_tile(lb + 0u * 2112u, wgt, ISDIM);
  stage_tile(lb + 1u * 2112u, wut, ISDIM);

  v8f aG[4] = {}, aU[4] = {};
  int buf = 0;
  for (int k0 = 0; k0 < HDIM; k0 += 32) {
    const int kn = (k0 + 32 < HDIM) ? k0 + 32 : 0;
    stage_tile(lb + (u32)((buf ^ 1) * 2 + 0) * 2112u, wgt + (size_t)kn * ISDIM, ISDIM);
    stage_tile(lb + (u32)((buf ^ 1) * 2 + 1) * 2112u, wut + (size_t)kn * ISDIM, ISDIM);
    wait_async<8>();
    v16bf Bg = bfrag(&stage[buf * 2 + 0][0]);
    v16bf Bu = bfrag(&stage[buf * 2 + 1][0]);
    const int ka = k0 + hi * 8;
#pragma unroll
    for (int s = 0; s < 4; ++s) {
      v16bf A = pack16(*(const v8bf*)(xrows[s] + ka), *(const v8bf*)(xrows[s] + ka + 16));
      aG[s] = wmma_bf16(A, Bg, aG[s]);
      aU[s] = wmma_bf16(A, Bu, aU[s]);
    }
    buf ^= 1;
  }
#pragma unroll
  for (int s = 0; s < 4; ++s)
#pragma unroll
    for (int v = 0; v < 8; ++v) {
      const int r = rb * 64 + s * 16 + v + 8 * hi;
      float g = aG[s][v], u = aU[s][v];
      hs[(size_t)r * ISDIM + n] = (__bf16)(g * fast_rcp(1.f + __expf(-g)) * u);
    }
}

// ---------------- kernel 7: shared down + gated combine -> out ----------------
__global__ __launch_bounds__(32)
void shared_down_combine_kernel(const __bf16* __restrict__ hs,
                                const float* __restrict__ ws_down,
                                const float* __restrict__ sgate,
                                const float* __restrict__ partial,
                                float* __restrict__ out) {
  __shared__ float stage[2][528];
  const int COLT = HDIM / 16;
  const int ct = blockIdx.x % COLT;
  const int rb = blockIdx.x / COLT;
  const int l  = threadIdx.x;
  const int hi = l >> 4;
  const int n  = ct * 16 + (l & 15);
  const float* wdt = ws_down + ct * 16;

  const __bf16* arows[4];
#pragma unroll
  for (int s = 0; s < 4; ++s)
    arows[s] = hs + (size_t)(rb * 64 + s * 16 + (l & 15)) * ISDIM;

  const u32 lb = (u32)(uintptr_t)&stage[0][0];
  stage_tile(lb, wdt, HDIM);

  v8f acc[4] = {};
  int buf = 0;
  for (int k0 = 0; k0 < ISDIM; k0 += 32) {
    const int kn = (k0 + 32 < ISDIM) ? k0 + 32 : 0;
    stage_tile(lb + (u32)(buf ^ 1) * 2112u, wdt + (size_t)kn * HDIM, HDIM);
    wait_async<4>();
    v16bf B = bfrag(&stage[buf][0]);
    const int ka = k0 + hi * 8;
#pragma unroll
    for (int s = 0; s < 4; ++s) {
      v16bf A = pack16(*(const v8bf*)(arows[s] + ka), *(const v8bf*)(arows[s] + ka + 16));
      acc[s] = wmma_bf16(A, B, acc[s]);
    }
    buf ^= 1;
  }
#pragma unroll
  for (int s = 0; s < 4; ++s)
#pragma unroll
    for (int v = 0; v < 8; ++v) {
      const int t = rb * 64 + s * 16 + v + 8 * hi;
      const float* pr = partial + (size_t)t * TOPK * HDIM + n;
      out[(size_t)t * HDIM + n] = acc[s][v] * sgate[t]
                                + pr[0] + pr[HDIM] + pr[2 * HDIM] + pr[3 * HDIM];
    }
}

// ---------------- workspace layout ----------------
static constexpr size_t OFF_XB    = 0;                                   // T*H bf16
static constexpr size_t OFF_HBUF  = OFF_XB   + (size_t)TTOK * HDIM * 2;  // E*T*I bf16
static constexpr size_t OFF_HS    = OFF_HBUF + (size_t)NEXP * TTOK * IDIM * 2; // T*IS bf16
static constexpr size_t OFF_PART  = OFF_HS   + (size_t)TTOK * ISDIM * 2; // T*4*H f32
static constexpr size_t OFF_TOPI  = OFF_PART + (size_t)TTOK * TOPK * HDIM * 4;
static constexpr size_t OFF_TOPW  = OFF_TOPI + (size_t)TTOK * TOPK * 4;
static constexpr size_t OFF_SG    = OFF_TOPW + (size_t)TTOK * TOPK * 4;
static constexpr size_t OFF_LISTS = OFF_SG   + (size_t)TTOK * 4;
static constexpr size_t OFF_CNT   = OFF_LISTS + (size_t)NEXP * TTOK * 4;

extern "C" void kernel_launch(void* const* d_in, const int* in_sizes, int n_in,
                              void* d_out, int out_size, void* d_ws, size_t ws_size,
                              hipStream_t stream) {
  (void)in_sizes; (void)n_in; (void)out_size; (void)ws_size;
  const float* x        = (const float*)d_in[0];
  const float* w_router = (const float*)d_in[1];
  const float* w_gate   = (const float*)d_in[2];
  const float* w_up     = (const float*)d_in[3];
  const float* w_down   = (const float*)d_in[4];
  const float* ws_gate  = (const float*)d_in[5];
  const float* ws_up    = (const float*)d_in[6];
  const float* ws_down  = (const float*)d_in[7];
  const float* w_sg     = (const float*)d_in[8];

  float* out        = (float*)d_out;                       // [T,H]
  float* logits_out = (float*)d_out + (size_t)TTOK * HDIM; // [T,E]

  char* ws = (char*)d_ws;
  __bf16* xb     = (__bf16*)(ws + OFF_XB);
  __bf16* hbuf   = (__bf16*)(ws + OFF_HBUF);
  __bf16* hs     = (__bf16*)(ws + OFF_HS);
  float*  part   = (float*) (ws + OFF_PART);
  int*    top_i  = (int*)   (ws + OFF_TOPI);
  float*  top_w  = (float*) (ws + OFF_TOPW);
  float*  sgate  = (float*) (ws + OFF_SG);
  int*    lists  = (int*)   (ws + OFF_LISTS);
  int*    counts = (int*)   (ws + OFF_CNT);

  convert_x_kernel<<<(TTOK * HDIM + 255) / 256, 256, 0, stream>>>(x, xb);
  router_kernel<<<TTOK, 256, 0, stream>>>(x, w_router, w_sg, logits_out,
                                          top_i, top_w, sgate);
  build_lists_kernel<<<1, NEXP, 0, stream>>>(top_i, lists, counts);

  expert_gateup_kernel<<<NEXP * (TTOK / 64) * (IDIM / 16), 32, 0, stream>>>(
      xb, w_gate, w_up, lists, counts, hbuf);
  expert_down_kernel<<<NEXP * (TTOK / 64) * (HDIM / 16), 32, 0, stream>>>(
      hbuf, w_down, lists, counts, top_w, part);

  shared_gateup_kernel<<<(TTOK / 64) * (ISDIM / 16), 32, 0, stream>>>(
      xb, ws_gate, ws_up, hs);
  shared_down_combine_kernel<<<(TTOK / 64) * (HDIM / 16), 32, 0, stream>>>(
      hs, ws_down, sgate, part, out);
}
